// Transformer_enrichment_83528523972974
// MI455X (gfx1250) — compile-verified, matrix operand order checked by reading, not run on hardware
//
#include <hip/hip_runtime.h>
#include <hip/hip_bf16.h>
#include <math.h>

// ---------------- types ----------------
typedef __attribute__((ext_vector_type(16))) __bf16       v16bf;
typedef __attribute__((ext_vector_type(8)))  float        v8f;
typedef __attribute__((ext_vector_type(8)))  unsigned int u8i;
typedef __attribute__((ext_vector_type(4)))  unsigned int u4i;
typedef __attribute__((ext_vector_type(2)))  unsigned int u2i;
typedef __attribute__((ext_vector_type(4)))  float        f4;

#define LN_EPS 1e-5f

// ---- float -> bf16 (prefer native packed convert on gfx1250) ----
__device__ __forceinline__ unsigned int f2bf1(float f) {
    unsigned int u = __float_as_uint(f);
    unsigned int r = u + 0x7FFFu + ((u >> 16) & 1u);
    return r >> 16;
}
__device__ __forceinline__ unsigned int pkbf(float a, float b) {
#if __has_builtin(__builtin_amdgcn_cvt_pk_bf16_f32)
    typedef __attribute__((ext_vector_type(2))) __bf16 v2bf;
    v2bf r = __builtin_amdgcn_cvt_pk_bf16_f32(a, b);
    return __builtin_bit_cast(unsigned int, r);
#else
    return f2bf1(a) | (f2bf1(b) << 16);
#endif
}
__device__ __forceinline__ unsigned short f2bfs(float a) {
    return (unsigned short)(pkbf(a, a) & 0xFFFFu);
}

// ---- WMMA fragment builders ----
// 16-element bf16 fragment from two contiguous 8-element bf16 runs (16B each).
__device__ __forceinline__ v16bf frag_from_bf16(const unsigned short* __restrict__ p0,
                                                const unsigned short* __restrict__ p1) {
    u4i a = *(const u4i*)p0;
    u4i b = *(const u4i*)p1;
    u8i w;
    w[0] = a[0]; w[1] = a[1]; w[2] = a[2]; w[3] = a[3];
    w[4] = b[0]; w[5] = b[1]; w[6] = b[2]; w[7] = b[3];
    return __builtin_bit_cast(v16bf, w);
}
// Same from two contiguous 8-float runs (for the attn matrix, which must stay f32).
__device__ __forceinline__ v16bf frag_from_f32(const float* __restrict__ p0,
                                               const float* __restrict__ p1) {
    f4 a = *(const f4*)p0;
    f4 b = *(const f4*)(p0 + 4);
    f4 c = *(const f4*)p1;
    f4 d = *(const f4*)(p1 + 4);
    u8i w;
    w[0] = pkbf(a[0], a[1]); w[1] = pkbf(a[2], a[3]);
    w[2] = pkbf(b[0], b[1]); w[3] = pkbf(b[2], b[3]);
    w[4] = pkbf(c[0], c[1]); w[5] = pkbf(c[2], c[3]);
    w[6] = pkbf(d[0], d[1]); w[7] = pkbf(d[2], d[3]);
    return __builtin_bit_cast(v16bf, w);
}
__device__ __forceinline__ v8f wmma_bf16(v16bf a, v16bf b, v8f c) {
    return __builtin_amdgcn_wmma_f32_16x16x32_bf16(false, a, false, b, (short)0, c,
                                                   false, false);
}
#define V8F_ZERO {0.f, 0.f, 0.f, 0.f, 0.f, 0.f, 0.f, 0.f}

// ---------------- weight convert: W[K][N] f32 -> WT[N][K] bf16 ----------------
__global__ void conv_wt(const float* __restrict__ W, unsigned short* __restrict__ WT,
                        int Hd) {
    int i = blockIdx.x * blockDim.x + threadIdx.x;
    if (i >= Hd * Hd) return;
    int k = i % Hd;
    int n = i / Hd;
    WT[i] = f2bfs(W[(size_t)k * Hd + n]);
}

// ---------------- activation convert: f32 -> bf16 (flat, vectorized) ----------
__global__ void conv_act(const float* __restrict__ src, unsigned short* __restrict__ dst,
                         size_t n4) {
    size_t i = (size_t)blockIdx.x * blockDim.x + threadIdx.x;
    if (i >= n4) return;
    f4 v = ((const f4*)src)[i];
    u2i o;
    o[0] = pkbf(v[0], v[1]);
    o[1] = pkbf(v[2], v[3]);
    ((u2i*)dst)[i] = o;
}

// ---------------- GEMM: C = A_bf16[M,K] @ WT[N,K]^T (+bias, opt ELU) ----------
// wave computes a 16x32 tile (2 WMMAs sharing the A fragment); block = 8 waves.
__global__ __launch_bounds__(256) void gemm_xw(const unsigned short* __restrict__ A,
                                               const unsigned short* __restrict__ WT,
                                               const float* __restrict__ bias,
                                               float* __restrict__ C,
                                               unsigned short* __restrict__ Cbf,
                                               int M, int N, int K, int act) {
    const int lane = threadIdx.x & 31;
    const int wave = threadIdx.x >> 5;
    const int lo = lane & 15, hi = lane >> 4;
    const int tileN = (blockIdx.x * 8 + wave) * 32;
    const int tileM = blockIdx.y * 16;

    const unsigned short* arow  = A  + (size_t)(tileM + lo) * K;
    const unsigned short* brow0 = WT + (size_t)(tileN + lo) * K;
    const unsigned short* brow1 = brow0 + (size_t)16 * K;

    v8f c0 = V8F_ZERO, c1 = V8F_ZERO;
    for (int kb = 0; kb < K; kb += 32) {
        __builtin_prefetch(arow + kb + 32, 0, 3);   // global_prefetch_b8, WGP scope
        const int o0 = kb + 8 * hi;
        const int o1 = kb + 16 + 8 * hi;
        v16bf af = frag_from_bf16(arow + o0, arow + o1);
        c0 = wmma_bf16(af, frag_from_bf16(brow0 + o0, brow0 + o1), c0);
        c1 = wmma_bf16(af, frag_from_bf16(brow1 + o0, brow1 + o1), c1);
    }
    const float bv0 = bias ? bias[tileN + lo] : 0.0f;
    const float bv1 = bias ? bias[tileN + 16 + lo] : 0.0f;
    const size_t base = (size_t)(tileM + 8 * hi) * N + tileN + lo;
#pragma unroll
    for (int r = 0; r < 8; ++r) {
        float v0 = c0[r] + bv0;
        float v1 = c1[r] + bv1;
        if (act == 1) {
            v0 = v0 > 0.f ? v0 : expm1f(v0);
            v1 = v1 > 0.f ? v1 : expm1f(v1);
        }
        if (C) {
            C[base + (size_t)r * N] = v0;
            C[base + (size_t)r * N + 16] = v1;
        }
        if (Cbf) {
            Cbf[base + (size_t)r * N] = f2bfs(v0);
            Cbf[base + (size_t)r * N + 16] = f2bfs(v1);
        }
    }
}

// ---------------- attention scores: masked Q K^T / sqrt(hd) -------------------
// wave computes one 16-query x 32-key tile (2 WMMAs sharing the Q fragment).
__global__ __launch_bounds__(256) void attn_scores(const unsigned short* __restrict__ Q,
                                                   const unsigned short* __restrict__ Kmat,
                                                   const int* __restrict__ lengths,
                                                   float* __restrict__ attn,
                                                   int T, int Hd, int nh, int hd) {
    const int lane = threadIdx.x & 31;
    const int wave = threadIdx.x >> 5;
    const int lo = lane & 15, hi = lane >> 4;
    const int kt0 = (blockIdx.x * 8 + wave) * 2;  // first of 2 key tiles
    const int qt = blockIdx.y;
    const int bh = blockIdx.z;
    const int b = bh / nh, h = bh % nh;

    const unsigned short* qrow  = Q    + ((size_t)b * T + qt * 16 + lo) * Hd + h * hd;
    const unsigned short* krow0 = Kmat + ((size_t)b * T + kt0 * 16 + lo) * Hd + h * hd;
    const unsigned short* krow1 = krow0 + (size_t)16 * Hd;

    v8f c0 = V8F_ZERO, c1 = V8F_ZERO;
    for (int kb = 0; kb < hd; kb += 32) {
        const int o0 = kb + 8 * hi;
        const int o1 = kb + 16 + 8 * hi;
        v16bf qf = frag_from_bf16(qrow + o0, qrow + o1);
        c0 = wmma_bf16(qf, frag_from_bf16(krow0 + o0, krow0 + o1), c0);
        c1 = wmma_bf16(qf, frag_from_bf16(krow1 + o0, krow1 + o1), c1);
    }
    const int len = lengths[b];
    const int kk0 = kt0 * 16 + lo;
    const int kk1 = kk0 + 16;
    const float scale = rsqrtf((float)hd);
    float* orow = attn + ((size_t)bh * T + qt * 16 + 8 * hi) * T;
#pragma unroll
    for (int r = 0; r < 8; ++r) {
        orow[(size_t)r * T + kk0] = (kk0 < len) ? c0[r] * scale : -1e10f;
        orow[(size_t)r * T + kk1] = (kk1 < len) ? c1[r] * scale : -1e10f;
    }
}

// ---------------- row softmax in place (T = 512, one wave per row) ------------
__global__ __launch_bounds__(256) void softmax_rows(float* __restrict__ attn, int T) {
    const int lane = threadIdx.x & 31;
    const int wave = threadIdx.x >> 5;
    const size_t row = (size_t)blockIdx.x * 8 + wave;
    float* p = attn + row * T;
    float v[16];
    float m = -3.0e38f;
#pragma unroll
    for (int j = 0; j < 16; ++j) { v[j] = p[lane + 32 * j]; m = fmaxf(m, v[j]); }
    for (int o = 16; o; o >>= 1) m = fmaxf(m, __shfl_xor(m, o, 32));
    float s = 0.f;
#pragma unroll
    for (int j = 0; j < 16; ++j) { v[j] = __expf(v[j] - m); s += v[j]; }
    for (int o = 16; o; o >>= 1) s += __shfl_xor(s, o, 32);
    const float inv = 1.0f / s;
#pragma unroll
    for (int j = 0; j < 16; ++j) p[lane + 32 * j] = v[j] * inv;
}

// ---------------- Vbf[B,T,H] -> VT[bh][d][t] (bf16 gather transpose) ----------
__global__ void build_vt(const unsigned short* __restrict__ V,
                         unsigned short* __restrict__ VT,
                         int T, int Hd, int nh, int hd, size_t n) {
    size_t i = (size_t)blockIdx.x * blockDim.x + threadIdx.x;
    if (i >= n) return;
    int t = (int)(i % T);
    size_t r = i / T;
    int d = (int)(r % hd);
    size_t bh = r / hd;
    int b = (int)(bh / nh), h = (int)(bh % nh);
    VT[i] = V[((size_t)b * T + t) * Hd + h * hd + d];
}

// ---------------- ctx_bf16 = (attn @ V) * sqrt(len) ---------------------------
// wave computes 16 t-rows x full head (4 WMMAs reuse one attn fragment).
__global__ __launch_bounds__(256) void attn_av(const float* __restrict__ attn,
                                               const unsigned short* __restrict__ VT,
                                               const int* __restrict__ lengths,
                                               unsigned short* __restrict__ ctxbf,
                                               int T, int Hd, int nh, int hd) {
    const int lane = threadIdx.x & 31;
    const int wave = threadIdx.x >> 5;
    const int lo = lane & 15, hi = lane >> 4;
    const int tt = blockIdx.x * 8 + wave;
    const int bh = blockIdx.y;
    const int b = bh / nh, h = bh % nh;

    const float* arow = attn + ((size_t)bh * T + tt * 16 + lo) * T;
    const unsigned short* bbase = VT + ((size_t)bh * hd + lo) * T;

    v8f c[4] = {V8F_ZERO, V8F_ZERO, V8F_ZERO, V8F_ZERO};
    for (int kb = 0; kb < T; kb += 32) {
        __builtin_prefetch(arow + kb + 32, 0, 3);
        const int o0 = kb + 8 * hi;
        const int o1 = kb + 16 + 8 * hi;
        v16bf af = frag_from_f32(arow + o0, arow + o1);
#pragma unroll
        for (int dt = 0; dt < 4; ++dt) {
            const unsigned short* brow = bbase + (size_t)dt * 16 * T;
            c[dt] = wmma_bf16(af, frag_from_bf16(brow + o0, brow + o1), c[dt]);
        }
    }
    const float scale = sqrtf((float)lengths[b]);
    unsigned short* crow =
        ctxbf + ((size_t)b * T + tt * 16 + 8 * hi) * Hd + h * hd + lo;
#pragma unroll
    for (int dt = 0; dt < 4; ++dt)
#pragma unroll
        for (int r = 0; r < 8; ++r)
            crow[(size_t)r * Hd + dt * 16] = f2bfs(c[dt][r] * scale);
}

// ---------------- fused GLU + LayerNorm (one wave per 512-wide row) -----------
// t = a*sigmoid(g) + r ; y = LN(t)*gamma + beta ; emit f32 and/or bf16
__global__ __launch_bounds__(256) void glu_ln(const float* __restrict__ a,
                                              const float* __restrict__ g,
                                              const float* __restrict__ r,
                                              const float* __restrict__ gamma,
                                              const float* __restrict__ beta,
                                              float* __restrict__ yf,
                                              unsigned short* __restrict__ ybf,
                                              int Hd) {
    const int lane = threadIdx.x & 31;
    const int wave = threadIdx.x >> 5;
    const size_t row = (size_t)blockIdx.x * 8 + wave;
    const size_t base = row * Hd;
    float v[16], s = 0.f, ss = 0.f;
#pragma unroll
    for (int j = 0; j < 16; ++j) {
        const int col = lane + 32 * j;
        const float av = a[base + col];
        const float gv = g[base + col];
        const float rv = r[base + col];
        v[j] = av * (1.0f / (1.0f + __expf(-gv))) + rv;
        s += v[j];
        ss += v[j] * v[j];
    }
    for (int o = 16; o; o >>= 1) { s += __shfl_xor(s, o, 32); ss += __shfl_xor(ss, o, 32); }
    const float inv = 1.0f / (float)Hd;
    const float mu = s * inv;
    const float var = ss * inv - mu * mu;
    const float rs = rsqrtf(var + LN_EPS);
#pragma unroll
    for (int j = 0; j < 16; ++j) {
        const int col = lane + 32 * j;
        const float y = (v[j] - mu) * rs * gamma[col] + beta[col];
        if (yf)  yf[base + col] = y;
        if (ybf) ybf[base + col] = f2bfs(y);
    }
}

// =============================== host launcher ================================
extern "C" void kernel_launch(void* const* d_in, const int* in_sizes, int n_in,
                              void* d_out, int out_size, void* d_ws, size_t ws_size,
                              hipStream_t stream) {
    const int Bb = 32, T = 512, Hd = 512, nh = 8, hd = Hd / nh;
    const size_t BTH = (size_t)Bb * T * Hd;          // 8,388,608
    (void)in_sizes; (void)n_in; (void)out_size; (void)ws_size;

    const float* x        = (const float*)d_in[0];
    const float* skip     = (const float*)d_in[1];
    /* d_in[2] = mask (derived from lengths) */
    const int*   lengths  = (const int*)d_in[3];
    const float* Wsrc[12] = {
        (const float*)d_in[4],  (const float*)d_in[5],  (const float*)d_in[6],
        (const float*)d_in[7],                                   // wq wk wv wo
        (const float*)d_in[8],  (const float*)d_in[10],          // glu_t_w, glu_t_wg
        (const float*)d_in[14], (const float*)d_in[16],          // grn_w1, grn_w2
        (const float*)d_in[18], (const float*)d_in[20],          // grn_glu_w, grn_glu_wg
        (const float*)d_in[24], (const float*)d_in[26]           // glu_f_w, glu_f_wg
    };
    const float* glu_t_b    = (const float*)d_in[9];
    const float* glu_t_bg   = (const float*)d_in[11];
    const float* ln_t_g     = (const float*)d_in[12];
    const float* ln_t_b     = (const float*)d_in[13];
    const float* grn_b1     = (const float*)d_in[15];
    const float* grn_b2     = (const float*)d_in[17];
    const float* grn_glu_b  = (const float*)d_in[19];
    const float* grn_glu_bg = (const float*)d_in[21];
    const float* grn_ln_g   = (const float*)d_in[22];
    const float* grn_ln_b   = (const float*)d_in[23];
    const float* glu_f_b    = (const float*)d_in[25];
    const float* glu_f_bg   = (const float*)d_in[27];
    const float* ln_f_g     = (const float*)d_in[28];
    const float* ln_f_b     = (const float*)d_in[29];

    float* out_x3 = (float*)d_out;
    float* attn   = (float*)d_out + BTH;             // [B, nh, T, T]

    // ws layout: WT bf16[12*H*H] | f32 fbuf0..2 [BTH] | bf16 bbuf0..3 [BTH]
    char* ws = (char*)d_ws;
    unsigned short* WT = (unsigned short*)ws;
    const size_t wbytes = 12ull * Hd * Hd * sizeof(unsigned short);
    float* fbase = (float*)(ws + wbytes);
    float* fbuf0 = fbase;                 // x1 (f32 residual)
    float* fbuf1 = fbase + BTH;           // A-branch of GLUs
    float* fbuf2 = fbase + 2 * BTH;       // G-branch of GLUs
    unsigned short* bbase = (unsigned short*)(fbase + 3 * BTH);
    unsigned short* bbuf0 = bbase;                 // xbf -> VT
    unsigned short* bbuf1 = bbase + BTH;           // Qbf -> Obf -> h2bf
    unsigned short* bbuf2 = bbase + 2 * BTH;       // Kbf -> x1bf
    unsigned short* bbuf3 = bbase + 3 * BTH;       // Vbf -> ctxbf -> Ebf -> x2bf

    // --- weights: f32 [K][N] -> bf16 [N][K] ---
    for (int i = 0; i < 12; ++i)
        conv_wt<<<(Hd * Hd + 255) / 256, 256, 0, stream>>>(
            Wsrc[i], WT + (size_t)i * Hd * Hd, Hd);

    const size_t n4 = BTH / 4;
    const unsigned cb = (unsigned)((n4 + 255) / 256);
    const unsigned lnb = (Bb * T) / 8;
    const dim3 gg(Hd / 256, (Bb * T) / 16);   // gemm grid: 32-col tiles x 8 waves

    auto run_gemm = [&](const unsigned short* A, int widx, const float* bias,
                        float* C, unsigned short* Cbf, int act) {
        gemm_xw<<<gg, 256, 0, stream>>>(A, WT + (size_t)widx * Hd * Hd, bias, C, Cbf,
                                        Bb * T, Hd, Hd, act);
    };

    // --- attention ---
    conv_act<<<cb, 256, 0, stream>>>(x, bbuf0, n4);          // xbf
    run_gemm(bbuf0, 0, nullptr, nullptr, bbuf1, 0);          // Qbf
    run_gemm(bbuf0, 1, nullptr, nullptr, bbuf2, 0);          // Kbf
    run_gemm(bbuf0, 2, nullptr, nullptr, bbuf3, 0);          // Vbf
    build_vt<<<(unsigned)((BTH + 255) / 256), 256, 0, stream>>>(
        bbuf3, bbuf0, T, Hd, nh, hd, BTH);                   // VT -> bbuf0
    attn_scores<<<dim3(T / 256, T / 16, Bb * nh), 256, 0, stream>>>(
        bbuf1, bbuf2, lengths, attn, T, Hd, nh, hd);
    softmax_rows<<<(Bb * nh * T) / 8, 256, 0, stream>>>(attn, T);
    attn_av<<<dim3(T / 16 / 8, Bb * nh), 256, 0, stream>>>(
        attn, bbuf0, lengths, bbuf3, T, Hd, nh, hd);         // ctxbf -> bbuf3
    run_gemm(bbuf3, 3, nullptr, nullptr, bbuf1, 0);          // Obf -> bbuf1

    // --- GLU_t + LN -> x1 (f32 fbuf0, bf16 bbuf2) ---
    run_gemm(bbuf1, 4, glu_t_b,  fbuf1, nullptr, 0);         // A1
    run_gemm(bbuf1, 5, glu_t_bg, fbuf2, nullptr, 0);         // G1
    glu_ln<<<lnb, 256, 0, stream>>>(fbuf1, fbuf2, x, ln_t_g, ln_t_b,
                                    fbuf0, bbuf2, Hd);

    // --- GRN ---
    run_gemm(bbuf2, 6, grn_b1, nullptr, bbuf3, 1);           // E = elu(x1 w1 + b1)
    run_gemm(bbuf3, 7, grn_b2, nullptr, bbuf1, 0);           // h2bf
    run_gemm(bbuf1, 8, grn_glu_b,  fbuf1, nullptr, 0);       // A2
    run_gemm(bbuf1, 9, grn_glu_bg, fbuf2, nullptr, 0);       // G2
    glu_ln<<<lnb, 256, 0, stream>>>(fbuf1, fbuf2, fbuf0, grn_ln_g, grn_ln_b,
                                    nullptr, bbuf3, Hd);     // x2bf -> bbuf3

    // --- final GLU + skip + LN -> x3 ---
    run_gemm(bbuf3, 10, glu_f_b,  fbuf1, nullptr, 0);        // A3
    run_gemm(bbuf3, 11, glu_f_bg, fbuf2, nullptr, 0);        // G3
    glu_ln<<<lnb, 256, 0, stream>>>(fbuf1, fbuf2, skip, ln_f_g, ln_f_b,
                                    out_x3, nullptr, Hd);
}
